// PiNet_74397423501703
// MI455X (gfx1250) — matrix operand order repaired and checked by reading, not compiled, not run on hardware
//
#include <hip/hip_runtime.h>
#include <hip/hip_bf16.h>

#define N_ATOMS 25000
#define N_PAIRS 500000
#define N_PROP  16
#define C       64
#define N_BASIS 4
#define DEPTH   4

typedef __attribute__((ext_vector_type(16))) _Float16 v16h;
typedef __attribute__((ext_vector_type(8)))  float    v8f;

// Branch-free tanh. Prefer the gfx1250 hardware transcendental (v_tanh_f32);
// otherwise use the exact identity tanh(x) = (e-1)/(e+1), e = exp2(2x*log2 e),
// via the 1-ulp HW exp2/rcp ops. Clamp keeps e finite (tanh(9) == 1.0f in f32).
__device__ __forceinline__ float fast_tanh(float x)
{
#if __has_builtin(__builtin_amdgcn_tanhf)
    return __builtin_amdgcn_tanhf(x);
#else
    float xc = fminf(9.0f, fmaxf(-9.0f, x));
    float e  = __builtin_amdgcn_exp2f(xc * 2.8853900817779268f); // 2*log2(e)
    return (e - 1.0f) * __builtin_amdgcn_rcpf(e + 1.0f);
#endif
}

// ---------------------------------------------------------------------------
// Weight prep: convert fp32 pair-stage weights (row-major K x N) into fp16
// WMMA B-fragment order. Fragment f = kt*NT + nt covers K in [kt*32,kt*32+32),
// N in [nt*16,nt*16+16). Per ISA layout: lane l holds column n = nt*16+(l&15),
// element j holds K = kt*32 + (l>>4)*8 + j (j<8), kt*32 + 16 + (l>>4)*8 + (j-8) (j>=8). Stored so each lane's 16 halves
// are contiguous (32B) -> two global_load_b128 per fragment.
// Regions (fp16 elems): pi_w1 [0,8192) pi_w2 [8192,24576)
//                       ii_w1 [24576,28672) ii_w2 [28672,32768)
// ---------------------------------------------------------------------------
__global__ __launch_bounds__(256) void prep_weights(
    const float* __restrict__ pi_w1, const float* __restrict__ pi_w2,
    const float* __restrict__ ii_w1, const float* __restrict__ ii_w2,
    _Float16* __restrict__ wf, int d)
{
    int gid = blockIdx.x * blockDim.x + threadIdx.x;   // 0..32767
    const float* src; int Ntot, NT; int idx = gid;
    if (idx < 8192)        { src = pi_w1 + (size_t)d * 128 * 64;  Ntot = 64;  NT = 4; }
    else if (idx < 24576)  { src = pi_w2 + (size_t)d * 64 * 256;  Ntot = 256; NT = 16; idx -= 8192; }
    else if (idx < 28672)  { src = ii_w1 + (size_t)d * 64 * 64;   Ntot = 64;  NT = 4;  idx -= 24576; }
    else                   { src = ii_w2 + (size_t)d * 64 * 64;   Ntot = 64;  NT = 4;  idx -= 28672; }
    int j    = idx & 15;
    int lane = (idx >> 4) & 31;
    int f    = idx >> 9;
    int kt   = f / NT, nt = f % NT;
    int k = kt * 32 + ((j < 8) ? ((lane >> 4) * 8 + j) : (16 + (lane >> 4) * 8 + (j - 8)));
    int n = nt * 16 + (lane & 15);
    wf[gid] = (_Float16)src[k * Ntot + n];
}

// ---------------------------------------------------------------------------
// Atom stage 1: h = tanh(tanh(x@W1+b1)@W2+b2); store fp16 h; zero newp.
// 4 atoms per 256-thread block; fp32 VALU (25k rows -> negligible cost).
// ---------------------------------------------------------------------------
__global__ __launch_bounds__(256) void atom_pp(
    const float* __restrict__ prop, const float* __restrict__ p_cur,
    const float* __restrict__ pp0_w1, const float* __restrict__ pp0_b1,
    const float* __restrict__ pp_w1,  const float* __restrict__ pp_b1,
    const float* __restrict__ pp_w2,  const float* __restrict__ pp_b2,
    _Float16* __restrict__ h_out, float* __restrict__ newp, int d)
{
    __shared__ float W1[64 * 64];
    __shared__ float W2[64 * 64];
    __shared__ float xrow[4][64];
    __shared__ float h1[4][64];
    int tid = threadIdx.x;
    int K1 = (d == 0) ? N_PROP : C;
    const float* w1 = (d == 0) ? pp0_w1 : pp_w1 + (size_t)(d - 1) * C * C;
    const float* b1 = (d == 0) ? pp0_b1 : pp_b1 + (size_t)(d - 1) * C;
    const float* w2 = pp_w2 + (size_t)d * C * C;
    const float* b2 = pp_b2 + (size_t)d * C;
    for (int i = tid; i < K1 * 64; i += 256) W1[i] = w1[i];
    for (int i = tid; i < 64 * 64; i += 256) W2[i] = w2[i];
    int a = tid >> 6, c = tid & 63;
    int atom = blockIdx.x * 4 + a;
    const float* x = (d == 0) ? prop : p_cur;
    int xw = (d == 0) ? N_PROP : C;
    if (c < xw) xrow[a][c] = x[(size_t)atom * xw + c];
    __syncthreads();
    float s = b1[c];
    for (int k = 0; k < K1; k++) s += xrow[a][k] * W1[k * 64 + c];
    h1[a][c] = fast_tanh(s);
    __syncthreads();
    float s2 = b2[c];
    for (int k = 0; k < 64; k++) s2 += h1[a][k] * W2[k * 64 + c];
    s2 = fast_tanh(s2);
    h_out[(size_t)atom * 64 + c] = (_Float16)s2;
    newp[(size_t)atom * 64 + c]  = 0.0f;
}

// ---------------------------------------------------------------------------
// Fused pair kernel (WMMA). 8 waves/block, each wave = one 16-pair M-tile.
// GEMM1 (K=128,N=64) -> tanh -> GEMM2 (K=64,N=256) -> tanh -> basis contract
// -> GEMM3 -> tanh -> GEMM4 -> tanh -> atomic segment-sum into newp.
// ---------------------------------------------------------------------------
__global__ __launch_bounds__(256) void pair_kernel(
    const int* __restrict__ ind2, const float* __restrict__ basis,
    const _Float16* __restrict__ h_in, const _Float16* __restrict__ wf,
    const float* __restrict__ pi_b1, const float* __restrict__ pi_b2,
    float* __restrict__ newp, int d)
{
    __shared__ float stage_all[8][16 * 64];      // per-wave 16x64 f32 tile
    const int tid  = threadIdx.x;
    const int wave = tid >> 5;
    const int lane = tid & 31;
    const int half = lane >> 4;                  // K-half selector (A/B layout)
    const int lm   = lane & 15;                  // row (A) / col (B,C,D) index
    float* stage = stage_all[wave];
    const long pbase = (long)(blockIdx.x * 8 + wave) * 16;

    // --- indices for this lane's A row (m = lm) ---
    long pa = pbase + lm; if (pa >= N_PAIRS) pa = N_PAIRS - 1;
    const int gi = ind2[pa * 2 + 0];
    const int gj = ind2[pa * 2 + 1];

    // --- A fragments for GEMM1: gather pij = [h[i], h[j]] (16 x 128 fp16) ---
    v16h a1[4];
    #pragma unroll
    for (int kt = 0; kt < 4; kt++) {
        const int row = (kt < 2) ? gi : gj;
        const _Float16* src = h_in + (long)row * 64 + (kt & 1) * 32;
        #pragma unroll
        for (int e = 0; e < 16; e++) {
            int K = (e < 8) ? (half * 8 + e) : (16 + half * 8 + (e - 8));
            a1[kt][e] = src[K];
        }
    }

    // --- GEMM1: (16x128)@(128x64), bias + tanh -> stage ---
    #pragma unroll
    for (int nt = 0; nt < 4; nt++) {
        v8f acc = {};
        #pragma unroll
        for (int kt = 0; kt < 4; kt++) {
            v16h b = *(const v16h*)(wf + ((kt * 4 + nt) * 32 + lane) * 16);
            acc = __builtin_amdgcn_wmma_f32_16x16x32_f16(false, a1[kt], false, b,
                                                         (short)0, acc, false, false);
        }
        float bias = pi_b1[d * 64 + nt * 16 + lm];
        #pragma unroll
        for (int r = 0; r < 8; r++)
            stage[(r + 8 * half) * 64 + nt * 16 + lm] = fast_tanh(acc[r] + bias);
    }

    // --- restage as A fragments (K=64 -> 2 fragments) ---
    v16h a2[2];
    #pragma unroll
    for (int kt = 0; kt < 2; kt++)
        #pragma unroll
        for (int e = 0; e < 16; e++) {
            int K = kt * 32 + ((e < 8) ? (half * 8 + e) : (16 + half * 8 + (e - 8)));
            a2[kt][e] = (_Float16)stage[lm * 64 + K];
        }

    // --- basis values: lane handles b = lane&3, rows m = r + 8*half ---
    float bas[8];
    #pragma unroll
    for (int r = 0; r < 8; r++) {
        long p = pbase + r + 8 * half; if (p >= N_PAIRS) p = N_PAIRS - 1;
        bas[r] = basis[p * 4 + (lane & 3)];
    }

    // --- GEMM2: (16x64)@(64x256), bias in accumulator, tanh, basis-contract ---
    const _Float16* wf2 = wf + 8192;
    const int cc = lm >> 2;                      // channel within nt group
    for (int nt = 0; nt < 16; nt++) {
        float bias = pi_b2[d * 256 + nt * 16 + lm];
        v8f acc;
        #pragma unroll
        for (int r = 0; r < 8; r++) acc[r] = bias;
        #pragma unroll
        for (int kt = 0; kt < 2; kt++) {
            v16h b = *(const v16h*)(wf2 + ((kt * 16 + nt) * 32 + lane) * 16);
            acc = __builtin_amdgcn_wmma_f32_16x16x32_f16(false, a2[kt], false, b,
                                                         (short)0, acc, false, false);
        }
        #pragma unroll
        for (int r = 0; r < 8; r++) {
            float v = fast_tanh(acc[r]) * bas[r];
            v += __shfl_xor(v, 1);               // reduce over basis index b
            v += __shfl_xor(v, 2);
            if ((lane & 3) == 0)
                stage[(r + 8 * half) * 64 + nt * 4 + cc] = v;   // inter[m][c]
        }
    }

    // --- GEMM3: inter @ ii_w1 (no bias), tanh ---
    v16h a3[2];
    #pragma unroll
    for (int kt = 0; kt < 2; kt++)
        #pragma unroll
        for (int e = 0; e < 16; e++) {
            int K = kt * 32 + ((e < 8) ? (half * 8 + e) : (16 + half * 8 + (e - 8)));
            a3[kt][e] = (_Float16)stage[lm * 64 + K];
        }
    const _Float16* wf3 = wf + 24576;
    #pragma unroll
    for (int nt = 0; nt < 4; nt++) {
        v8f acc = {};
        #pragma unroll
        for (int kt = 0; kt < 2; kt++) {
            v16h b = *(const v16h*)(wf3 + ((kt * 4 + nt) * 32 + lane) * 16);
            acc = __builtin_amdgcn_wmma_f32_16x16x32_f16(false, a3[kt], false, b,
                                                         (short)0, acc, false, false);
        }
        #pragma unroll
        for (int r = 0; r < 8; r++)
            stage[(r + 8 * half) * 64 + nt * 16 + lm] = fast_tanh(acc[r]);
    }

    // --- GEMM4: @ ii_w2, tanh, atomic segment-sum into newp[ind_i] ---
    v16h a4[2];
    #pragma unroll
    for (int kt = 0; kt < 2; kt++)
        #pragma unroll
        for (int e = 0; e < 16; e++) {
            int K = kt * 32 + ((e < 8) ? (half * 8 + e) : (16 + half * 8 + (e - 8)));
            a4[kt][e] = (_Float16)stage[lm * 64 + K];
        }
    int  ai[8]; bool av[8];
    #pragma unroll
    for (int r = 0; r < 8; r++) {
        long p = pbase + r + 8 * half;
        av[r] = (p < N_PAIRS);
        ai[r] = ind2[(av[r] ? p : 0) * 2];
    }
    const _Float16* wf4 = wf + 28672;
    #pragma unroll
    for (int nt = 0; nt < 4; nt++) {
        v8f acc = {};
        #pragma unroll
        for (int kt = 0; kt < 2; kt++) {
            v16h b = *(const v16h*)(wf4 + ((kt * 4 + nt) * 32 + lane) * 16);
            acc = __builtin_amdgcn_wmma_f32_16x16x32_f16(false, a4[kt], false, b,
                                                         (short)0, acc, false, false);
        }
        #pragma unroll
        for (int r = 0; r < 8; r++)
            if (av[r])
                atomicAdd(&newp[(long)ai[r] * 64 + nt * 16 + lm], fast_tanh(acc[r]));
    }
}

// ---------------------------------------------------------------------------
// Atom stage 2: residual update + output head; accumulate d_out.
// ---------------------------------------------------------------------------
__global__ __launch_bounds__(256) void atom_update(
    const float* __restrict__ prop, float* __restrict__ p_cur,
    const float* __restrict__ newp, const float* __restrict__ res0_w,
    const float* __restrict__ out_w1, const float* __restrict__ out_b1,
    const float* __restrict__ out_w2, const float* __restrict__ out_b2,
    const float* __restrict__ out_wo, float* __restrict__ out, int d)
{
    __shared__ float W1[64 * 64];
    __shared__ float W2[64 * 64];
    __shared__ float R0[16 * 64];
    __shared__ float prow[4][64];
    __shared__ float o1[4][64];
    __shared__ float o2[4][64];
    int tid = threadIdx.x;
    for (int i = tid; i < 64 * 64; i += 256) {
        W1[i] = out_w1[(size_t)d * 4096 + i];
        W2[i] = out_w2[(size_t)d * 4096 + i];
    }
    if (d == 0)
        for (int i = tid; i < 16 * 64; i += 256) R0[i] = res0_w[i];
    int a = tid >> 6, c = tid & 63;
    int atom = blockIdx.x * 4 + a;
    __syncthreads();
    float pv;
    if (d == 0) {
        float s = newp[(size_t)atom * 64 + c];
        for (int k = 0; k < 16; k++) s += prop[(size_t)atom * 16 + k] * R0[k * 64 + c];
        pv = s;
    } else {
        pv = p_cur[(size_t)atom * 64 + c] + newp[(size_t)atom * 64 + c];
    }
    p_cur[(size_t)atom * 64 + c] = pv;
    prow[a][c] = pv;
    __syncthreads();
    float s1 = out_b1[d * 64 + c];
    for (int k = 0; k < 64; k++) s1 += prow[a][k] * W1[k * 64 + c];
    o1[a][c] = fast_tanh(s1);
    __syncthreads();
    float s2 = out_b2[d * 64 + c];
    for (int k = 0; k < 64; k++) s2 += o1[a][k] * W2[k * 64 + c];
    o2[a][c] = fast_tanh(s2);
    __syncthreads();
    if (c == 0) {
        float s = 0.0f;
        for (int k = 0; k < 64; k++) s += o2[a][k] * out_wo[d * 64 + k];
        if (d == 0) out[atom] = s; else out[atom] += s;
    }
}

// ---------------------------------------------------------------------------
extern "C" void kernel_launch(void* const* d_in, const int* in_sizes, int n_in,
                              void* d_out, int out_size, void* d_ws, size_t ws_size,
                              hipStream_t stream)
{
    const int*   ind2   = (const int*)  d_in[0];
    const float* prop   = (const float*)d_in[1];
    const float* basis  = (const float*)d_in[2];
    const float* pp0_w1 = (const float*)d_in[3];
    const float* pp0_b1 = (const float*)d_in[4];
    const float* pp_w1  = (const float*)d_in[5];
    const float* pp_b1  = (const float*)d_in[6];
    const float* pp_w2  = (const float*)d_in[7];
    const float* pp_b2  = (const float*)d_in[8];
    const float* pi_w1  = (const float*)d_in[9];
    const float* pi_b1  = (const float*)d_in[10];
    const float* pi_w2  = (const float*)d_in[11];
    const float* pi_b2  = (const float*)d_in[12];
    const float* ii_w1  = (const float*)d_in[13];
    const float* ii_w2  = (const float*)d_in[14];
    const float* res0_w = (const float*)d_in[15];
    const float* out_w1 = (const float*)d_in[16];
    const float* out_b1 = (const float*)d_in[17];
    const float* out_w2 = (const float*)d_in[18];
    const float* out_b2 = (const float*)d_in[19];
    const float* out_wo = (const float*)d_in[20];
    float* out = (float*)d_out;

    char* ws = (char*)d_ws;
    float*    p_cur = (float*)ws;    ws += (size_t)N_ATOMS * 64 * sizeof(float);
    _Float16* h_f16 = (_Float16*)ws; ws += (size_t)N_ATOMS * 64 * sizeof(_Float16);
    float*    newp  = (float*)ws;    ws += (size_t)N_ATOMS * 64 * sizeof(float);
    _Float16* wfrag = (_Float16*)ws; // 32768 fp16 elems

    for (int d = 0; d < DEPTH; d++) {
        prep_weights<<<128, 256, 0, stream>>>(pi_w1, pi_w2, ii_w1, ii_w2, wfrag, d);
        atom_pp<<<N_ATOMS / 4, 256, 0, stream>>>(prop, p_cur, pp0_w1, pp0_b1,
                                                 pp_w1, pp_b1, pp_w2, pp_b2,
                                                 h_f16, newp, d);
        pair_kernel<<<(N_PAIRS + 127) / 128, 256, 0, stream>>>(ind2, basis, h_f16,
                                                               wfrag, pi_b1, pi_b2,
                                                               newp, d);
        atom_update<<<N_ATOMS / 4, 256, 0, stream>>>(prop, p_cur, newp, res0_w,
                                                     out_w1, out_b1, out_w2, out_b2,
                                                     out_wo, out, d);
    }
}
